// MessagePassingConvolution_40922448396306
// MI455X (gfx1250) — matrix operand order repaired and checked by reading, not compiled, not run on hardware
//
#include <hip/hip_runtime.h>
#include <hip/hip_bf16.h>

// ---------------------------------------------------------------------------
// MessagePassingConvolution for MI455X (gfx1250, wave32)
//  - 4-layer edge MLP (8->64->64->64->256) on v_wmma_f32_16x16x32_bf16
//  - weights repacked once per launch into WMMA B-fragment layout in d_ws
//  - equivariant tensor product + scatter via global_atomic_add_f32 (L2-resident)
// ---------------------------------------------------------------------------

typedef __attribute__((ext_vector_type(16))) __bf16          v16bf;
typedef __attribute__((ext_vector_type(8)))  float           v8f;

#define WPB   4          // waves per block (128 threads)
#define HSTR  73         // LDS stride for 16x64 activation tiles (gcd(73,64)=1)
#define MSTR  261        // LDS stride for 16x256 mix tile       (gcd(261,64)=1)

// Fast silu: one v_rcp_f32 instead of the IEEE divide expansion.
__device__ __forceinline__ float silu(float x) {
  return x * __builtin_amdgcn_rcpf(1.0f + __expf(-x));
}

__device__ __forceinline__ void atomic_add_f32(float* p, float v) {
  __hip_atomic_fetch_add(p, v, __ATOMIC_RELAXED, __HIP_MEMORY_SCOPE_AGENT);
}

// B-fragment: stored per lane as 16 contiguous bf16 (32B) -> one vector load.
__device__ __forceinline__ v16bf load_bfrag(const __bf16* __restrict__ ws,
                                            int frag, int lane) {
  return *reinterpret_cast<const v16bf*>(ws + (size_t)frag * 512 + (size_t)lane * 16);
}

// A-fragment (16x32 bf16) built from an LDS f32 tile [16][HSTR].
// Contiguous-k layout (matches B repack): lane L, half j -> M = L%16, K = (L/16)*16 + j
__device__ __forceinline__ v16bf build_a(const float* __restrict__ h, int kbase, int lane) {
  const int M  = lane & 15;
  const int hh = lane >> 4;
  v16bf t;
#pragma unroll
  for (int j = 0; j < 16; ++j) {
    const int k = kbase + (hh << 4) + j;
    t[j] = (__bf16)h[M * HSTR + k];          // native v_cvt_*_bf16_f32
  }
  return t;
}

// C/D layout: VGPR r, lane L -> M = r + (L/16)*8, N = L%16
__device__ __forceinline__ void store_tile(float* __restrict__ h, int stride, int nt,
                                           int lane, v8f acc, bool act) {
  const int N  = nt * 16 + (lane & 15);
  const int Mb = (lane >> 4) << 3;
#pragma unroll
  for (int r = 0; r < 8; ++r) {
    float x = acc[r];
    if (act) x = silu(x);
    h[(Mb + r) * stride + N] = x;
  }
}

// 64->64 layer: two K-chunks of 32, four N-tiles of 16, silu output.
__device__ __forceinline__ void mlp_layer64(const float* __restrict__ src,
                                            float* __restrict__ dst,
                                            const __bf16* __restrict__ ws,
                                            int fbase, int lane) {
  v16bf a0 = build_a(src, 0, lane);
  v16bf a1 = build_a(src, 32, lane);
#pragma unroll
  for (int nt = 0; nt < 4; ++nt) {
    v8f acc = {};
    acc = __builtin_amdgcn_wmma_f32_16x16x32_bf16(false, a0, false,
            load_bfrag(ws, fbase + nt, lane), (short)0, acc, false, false);
    acc = __builtin_amdgcn_wmma_f32_16x16x32_bf16(false, a1, false,
            load_bfrag(ws, fbase + 4 + nt, lane), (short)0, acc, false, false);
    store_tile(dst, HSTR, nt, lane, acc, true);
  }
}

// ---------------------------------------------------------------------------
// Kernel 0: zero the (poisoned) output accumulator.
// ---------------------------------------------------------------------------
__global__ void mpc_zero_out(float* __restrict__ out, int n4) {
  int i = blockIdx.x * blockDim.x + threadIdx.x;
  if (i < n4) reinterpret_cast<float4*>(out)[i] = make_float4(0.f, 0.f, 0.f, 0.f);
}

// ---------------------------------------------------------------------------
// Kernel 1: repack MLP weights into WMMA B-fragments (bf16) in d_ws.
// Fragment table (each fragment = 32 lanes x 16 bf16 = 512 elements):
//   L0: f=0..3    (K=8 zero-padded to 32, nt=0..3), scale 1/sqrt(8)
//   L1: f=4..11   (kc*4+nt),                        scale 1/8
//   L2: f=12..19  (kc*4+nt),                        scale 1/8
//   L3: f=20..51  (kc*16+nt),                       scale (1/8)*(1/sqrt(12))
// B layout: lane L, half j -> N = nt*16 + L%16, K = kc*32 + (L/16)*16 + j
// ---------------------------------------------------------------------------
__global__ void mpc_repack_weights(const float* __restrict__ W0,
                                   const float* __restrict__ W1,
                                   const float* __restrict__ W2,
                                   const float* __restrict__ W3,
                                   __bf16* __restrict__ ws) {
  const int f = blockIdx.x;
  const int lane = threadIdx.x;
  int layer, kc, nt;
  if (f < 4)       { layer = 0; kc = 0;            nt = f; }
  else if (f < 12) { layer = 1; kc = (f - 4) >> 2;  nt = (f - 4) & 3; }
  else if (f < 20) { layer = 2; kc = (f - 12) >> 2; nt = (f - 12) & 3; }
  else             { layer = 3; kc = (f - 20) >> 4; nt = (f - 20) & 15; }

  const float* W; int Kdim, Ndim; float scale;
  switch (layer) {
    case 0:  W = W0; Kdim = 8;  Ndim = 64;  scale = 0.35355339f;               break;
    case 1:  W = W1; Kdim = 64; Ndim = 64;  scale = 0.125f;                    break;
    case 2:  W = W2; Kdim = 64; Ndim = 64;  scale = 0.125f;                    break;
    default: W = W3; Kdim = 64; Ndim = 256; scale = 0.125f * 0.28867513f;      break;
  }

  const int n  = nt * 16 + (lane & 15);
  const int kb = kc * 32 + ((lane >> 4) << 4);
  __bf16* dst = ws + (size_t)f * 512 + (size_t)lane * 16;
#pragma unroll
  for (int j = 0; j < 16; ++j) {
    const int k = kb + j;
    const float v = (k < Kdim) ? W[k * Ndim + n] * scale : 0.0f;
    dst[j] = (__bf16)v;
  }
}

// ---------------------------------------------------------------------------
// Kernel 2: edge kernel. One wave owns 16 edges (M-tile). 128 threads/block.
// ---------------------------------------------------------------------------
__global__ void __launch_bounds__(128)
mpc_edge_kernel(const float* __restrict__ node_s,     // [N,64]
                const float* __restrict__ node_v,     // [N,64,3]
                const float* __restrict__ edge_s,     // [E,8]
                const float* __restrict__ edge_sh,    // [E,3]
                const int*   __restrict__ senders,    // [E]
                const int*   __restrict__ receivers,  // [E]
                const __bf16* __restrict__ ws,
                float* __restrict__ out,              // [N,512]
                int E) {
  __shared__ float hA[WPB][16][HSTR];
  __shared__ float hB[WPB][16][HSTR];
  __shared__ float mixb[WPB][16][MSTR];

  const int wave = threadIdx.x >> 5;
  const int lane = threadIdx.x & 31;
  const int eBase = (blockIdx.x * WPB + wave) * 16;
  if (eBase + 16 > E) return;   // E % 64 == 0 here; guard anyway (wave-uniform)

  const int eL   = lane & 15;
  const int half = lane >> 4;
  const int e    = eBase + eL;
  const int snd  = senders[e];

  // Warm L2/L0 with the sender rows while the WMMA MLP runs.
  const float* Srow = node_s + (size_t)snd * 64 + half * 32;
  const float* Vrow = node_v + (size_t)snd * 192 + (size_t)half * 96;
  __builtin_prefetch(Srow, 0, 1);
  __builtin_prefetch(Vrow, 0, 1);
  __builtin_prefetch(Vrow + 32, 0, 1);
  __builtin_prefetch(Vrow + 64, 0, 1);

  float* hAw = &hA[wave][0][0];
  float* hBw = &hB[wave][0][0];
  float* mxw = &mixb[wave][0][0];

  // ---- Layer 0: A from edge_scalars (K=8 valid, rest zero), 8 -> 64 ----
  {
    v16bf a0;
    const float* es = edge_s + (size_t)(eBase + eL) * 8;
#pragma unroll
    for (int j = 0; j < 16; ++j) {
      float v = 0.0f;
      if (half == 0 && j < 8) v = es[j];   // lane<16, K=j<8; everything else is 0
      a0[j] = (__bf16)v;
    }
#pragma unroll
    for (int nt = 0; nt < 4; ++nt) {
      v8f acc = {};
      acc = __builtin_amdgcn_wmma_f32_16x16x32_bf16(false, a0, false,
              load_bfrag(ws, nt, lane), (short)0, acc, false, false);
      store_tile(hAw, HSTR, nt, lane, acc, true);
    }
  }
  __syncthreads();

  // ---- Layers 1, 2: 64 -> 64 ----
  mlp_layer64(hAw, hBw, ws, 4, lane);
  __syncthreads();
  mlp_layer64(hBw, hAw, ws, 12, lane);
  __syncthreads();

  // ---- Layer 3: 64 -> 256 (mix), no activation; 1/sqrt(12) folded in ----
  {
    v16bf a0 = build_a(hAw, 0, lane);
    v16bf a1 = build_a(hAw, 32, lane);
#pragma unroll
    for (int nt = 0; nt < 16; ++nt) {
      v8f acc = {};
      acc = __builtin_amdgcn_wmma_f32_16x16x32_bf16(false, a0, false,
              load_bfrag(ws, 20 + nt, lane), (short)0, acc, false, false);
      acc = __builtin_amdgcn_wmma_f32_16x16x32_bf16(false, a1, false,
              load_bfrag(ws, 36 + nt, lane), (short)0, acc, false, false);
      store_tile(mxw, MSTR, nt, lane, acc, false);
    }
  }
  __syncthreads();

  // ---- Equivariant tensor product + scatter-add ----
  // Lane pair (L, L+16) shares edge eL; each lane handles 32 of 64 channels.
  const int rcv = receivers[e];
  const float sh0 = edge_sh[(size_t)e * 3 + 0];
  const float sh1 = edge_sh[(size_t)e * 3 + 1];
  const float sh2 = edge_sh[(size_t)e * 3 + 2];
  float* O = out + (size_t)rcv * 512;
  const float* mixe = mxw + eL * MSTR;
  const int cb = half * 32;
  const float inv_sqrt3 = 0.57735027f;

#pragma unroll 4
  for (int c0 = 0; c0 < 32; ++c0) {
    const int c = cb + c0;
    const float s  = Srow[c0];
    const float v0 = Vrow[c0 * 3 + 0];
    const float v1 = Vrow[c0 * 3 + 1];
    const float v2 = Vrow[c0 * 3 + 2];
    const float ms = mixe[c];          // mix for scalar channel c       (msg_s[0:64])
    const float mt = mixe[64 + c];     // mix for tp_0e channel          (msg_s[64:128])
    const float mv = mixe[128 + c];    // mix for vector channel         (msg_v[0:64])
    const float mp = mixe[192 + c];    // mix for tp_1o channel          (msg_v[64:128])

    const float tp0 = (v0 * sh0 + v1 * sh1 + v2 * sh2) * inv_sqrt3;

    atomic_add_f32(O + c,        s  * ms);
    atomic_add_f32(O + 64 + c,   tp0 * mt);

    float* Ov = O + 128 + c * 3;            // out_v channel c
    atomic_add_f32(Ov + 0, v0 * mv);
    atomic_add_f32(Ov + 1, v1 * mv);
    atomic_add_f32(Ov + 2, v2 * mv);

    float* Op = O + 128 + (64 + c) * 3;     // out_v channel 64+c (tp_1o)
    const float sp = s * mp;
    atomic_add_f32(Op + 0, sp * sh0);
    atomic_add_f32(Op + 1, sp * sh1);
    atomic_add_f32(Op + 2, sp * sh2);
  }
}

// ---------------------------------------------------------------------------
// Launch
// ---------------------------------------------------------------------------
extern "C" void kernel_launch(void* const* d_in, const int* in_sizes, int n_in,
                              void* d_out, int out_size, void* d_ws, size_t ws_size,
                              hipStream_t stream) {
  const float* node_s  = (const float*)d_in[0];
  const float* node_v  = (const float*)d_in[1];
  const float* edge_s  = (const float*)d_in[2];
  const float* edge_sh = (const float*)d_in[3];
  const int*   senders = (const int*)d_in[4];
  const int*   recv    = (const int*)d_in[5];
  const float* W0      = (const float*)d_in[6];
  const float* W1      = (const float*)d_in[7];
  const float* W2      = (const float*)d_in[8];
  const float* W3      = (const float*)d_in[9];
  float* out = (float*)d_out;
  __bf16* wsb = (__bf16*)d_ws;   // needs 52*512*2 = 53,248 bytes

  const int E = in_sizes[4];

  // 1) zero output accumulator (harness poisons d_out)
  const int n4 = out_size / 4;
  mpc_zero_out<<<(n4 + 255) / 256, 256, 0, stream>>>(out, n4);

  // 2) repack weights into WMMA B-fragments (52 fragments, one per block)
  mpc_repack_weights<<<52, 32, 0, stream>>>(W0, W1, W2, W3, wsb);

  // 3) edge kernel: 16 edges per wave, 4 waves per block
  const int blocks = (E + WPB * 16 - 1) / (WPB * 16);
  mpc_edge_kernel<<<blocks, 128, 0, stream>>>(node_s, node_v, edge_s, edge_sh,
                                              senders, recv, wsb, out, E);
}
